// GRU_Cell_41918880809222
// MI455X (gfx1250) — 8-bit matrix operand order repaired and checked by reading, compiled, not run on hardware
//
#include <hip/hip_runtime.h>

#define HID    128
#define TSTEPS 28
#define INF    28
#define WAVES  8
#define MTILE  (WAVES * 16)

typedef float v8f  __attribute__((ext_vector_type(8)));
typedef int   v8i  __attribute__((ext_vector_type(8)));
typedef int   v16i __attribute__((ext_vector_type(16)));

__device__ __forceinline__ unsigned int f32_to_fp8(float v) {
#if __has_builtin(__builtin_amdgcn_cvt_pk_fp8_f32)
    return (unsigned)__builtin_amdgcn_cvt_pk_fp8_f32(v, v, 0, false) & 0xFFu;
#else
    unsigned u = __float_as_uint(v);
    unsigned s = (u >> 24) & 0x80u;
    float a = fabsf(v);
    if (a < 0x1p-10f) return s;
    if (a > 448.0f) a = 448.0f;
    unsigned ua = __float_as_uint(a);
    int e0 = (int)(ua >> 23) - 127;
    unsigned m = (ua & 0x7FFFFFu) | 0x800000u;
    unsigned enc;
    if (e0 < -6) {
        int shift = 20 + (-6 - e0);
        unsigned q   = m >> shift;
        unsigned rem = m & ((1u << shift) - 1u);
        unsigned hlf = 1u << (shift - 1);
        q += (rem > hlf) || (rem == hlf && (q & 1u));
        enc = q;
    } else {
        unsigned q   = m >> 20;
        unsigned rem = m & 0xFFFFFu;
        q += (rem > 0x80000u) || (rem == 0x80000u && (q & 1u));
        enc = (((unsigned)(e0 + 7)) << 3) + q - 8u;
    }
    if (enc > 0x7Eu) enc = 0x7Eu;
    return s | enc;
#endif
}

__device__ __forceinline__ float tanh_(float x) {
#if __has_builtin(__builtin_amdgcn_tanhf)
    return __builtin_amdgcn_tanhf(x);
#else
    float e = __builtin_amdgcn_exp2f(x * -2.88539008f);
    return __builtin_amdgcn_rcpf(1.0f + e) * 2.0f - 1.0f;
#endif
}
__device__ __forceinline__ float sigm(float x) {
#if __has_builtin(__builtin_amdgcn_tanhf)
    return fmaf(0.5f, __builtin_amdgcn_tanhf(0.5f * x), 0.5f);
#else
    float e = __builtin_amdgcn_exp2f(x * -1.44269504f);
    return __builtin_amdgcn_rcpf(1.0f + e);
#endif
}

__device__ __forceinline__ v8f wmma64(v8i a, v8i b, v8f c) {
    return __builtin_amdgcn_wmma_f32_16x16x64_fp8_fp8(a, b, (short)0, c, false, false);
}
__device__ __forceinline__ v8f wmma128(v16i a, v16i b, v8f c) {
    return __builtin_amdgcn_wmma_f32_16x16x128_fp8_fp8(a, b, (short)0, c, false, false);
}

__device__ __forceinline__ v8i loadB64(const unsigned char* p, int lane) {
    uint4 a = *(const uint4*)(p + lane * 16);
    uint4 b = *(const uint4*)(p + 512 + lane * 16);
    v8i r;
    r[0]=(int)a.x; r[1]=(int)a.y; r[2]=(int)a.z; r[3]=(int)a.w;
    r[4]=(int)b.x; r[5]=(int)b.y; r[6]=(int)b.z; r[7]=(int)b.w;
    return r;
}
__device__ __forceinline__ v16i loadB128(const unsigned char* p, int lane) {
    v16i r;
#pragma unroll
    for (int q = 0; q < 4; ++q) {
        uint4 a = *(const uint4*)(p + q * 512 + lane * 16);
        r[4*q+0]=(int)a.x; r[4*q+1]=(int)a.y; r[4*q+2]=(int)a.z; r[4*q+3]=(int)a.w;
    }
    return r;
}

#define W1X_OFF 0
#define W1H_OFF 24576
#define W2X_OFF (24576 + 49152)
#define W2H_OFF (24576 + 2 * 49152)
#define WBYTES  (24576 + 3 * 49152)

__global__ void pack_weights_kernel(const float* __restrict__ w_ih1,
                                    const float* __restrict__ w_hh1,
                                    const float* __restrict__ w_ih2,
                                    const float* __restrict__ w_hh2,
                                    const float* __restrict__ b_ih1,
                                    const float* __restrict__ b_hh1,
                                    unsigned char* __restrict__ ws) {
    int idx = blockIdx.x * 256 + threadIdx.x;
    if (idx >= WBYTES) return;
    unsigned val8;
    if (idx < 24576) {
        int tile = idx >> 10;
        int r    = idx & 1023;
        int q    = r >> 9;
        int lane = (r >> 4) & 31;
        int bl   = r & 15;
        int v    = q * 4 + (bl >> 2);
        int b    = bl & 3;
        int K    = (v >> 2) * 32 + ((lane >= 16) ? 16 : 0) + (v & 3) * 4 + b;
        int col  = tile * 16 + (lane & 15);
        float w;
        if (K < INF)       w = w_ih1[col * INF + K];
        else if (K == INF) w = (col < 256) ? (b_ih1[col] + b_hh1[col])
                                           : b_ih1[col];
        else               w = 0.0f;
        val8 = f32_to_fp8(w);
    } else {
        int t   = idx - 24576;
        int mat = t / 49152;
        int loc = t % 49152;
        const float* W = (mat == 0) ? w_hh1 : ((mat == 1) ? w_ih2 : w_hh2);
        int tile = loc >> 11;
        int r    = loc & 2047;
        int q    = r >> 9;
        int rr   = r & 511;
        int lane = rr >> 4;
        int bl   = rr & 15;
        int v    = q * 4 + (bl >> 2);
        int b    = bl & 3;
        int K    = (v >> 2) * 32 + ((lane >= 16) ? 16 : 0) + (v & 3) * 4 + b;
        int col  = tile * 16 + (lane & 15);
        val8 = f32_to_fp8(W[col * HID + K]);
    }
    ws[idx] = (unsigned char)val8;
}

#define XPITCH 72
#define HPITCH 136

__global__ __launch_bounds__(256)
void gru_main(const float* __restrict__ x,
              const float* __restrict__ b_ih1, const float* __restrict__ b_hh1,
              const float* __restrict__ b_ih2, const float* __restrict__ b_hh2,
              const float* __restrict__ w_out, const float* __restrict__ b_out,
              const unsigned char* __restrict__ wsrc,
              float* __restrict__ out) {
    __shared__ __align__(16) unsigned char ldsw[WBYTES];
    __shared__ __align__(16) unsigned char xbuf[WAVES * 16 * XPITCH];
    __shared__ __align__(16) unsigned char hbuf[WAVES * 16 * HPITCH];
    __shared__ float lbias1n[128];
    __shared__ float lbias2[512];

    const int tid     = threadIdx.x;
    const int wave    = tid >> 5;
    const int lane    = tid & 31;
    const int rowBase = blockIdx.x * MTILE + wave * 16;
    const int colc    = lane & 15;
    const int hiHalf  = (lane >= 16) ? 1 : 0;

    {
        const uint4* g = (const uint4*)wsrc;
        uint4* s = (uint4*)ldsw;
        for (int i = tid; i < WBYTES / 16; i += 256) s[i] = g[i];
    }
    lbias2[tid] = b_ih2[tid] + b_hh2[tid];
    if (tid < 128) {
        lbias1n[tid]      = b_hh1[256 + tid];
        lbias2[256 + tid] = b_ih2[256 + tid];
        lbias2[384 + tid] = b_hh2[256 + tid];
    }
    for (int i = tid; i < (WAVES * 16 * XPITCH) / 4; i += 256)
        ((unsigned*)xbuf)[i] = 0u;
    for (int r = tid; r < WAVES * 16; r += 256)
        xbuf[r * XPITCH + INF] = 0x38;
    __syncthreads();

    unsigned char* myx = xbuf + wave * (16 * XPITCH);
    unsigned char* myh = hbuf + wave * (16 * HPITCH);

    float h1s[64], h2s[64];
    v16i ah1, ah2;
#pragma unroll
    for (int i = 0; i < 64; ++i) { h1s[i] = 0.f; h2s[i] = 0.f; }
#pragma unroll
    for (int i = 0; i < 16; ++i) { ah1[i] = 0; ah2[i] = 0; }

    for (int t = 0; t < TSTEPS; ++t) {
        if (t + 1 < TSTEPS) {
            const char* nx = (const char*)&x[(size_t)(rowBase + colc) * 784 + (t + 1) * INF];
            __builtin_prefetch(nx, 0, 1);
            __builtin_prefetch(nx + 64, 0, 1);
        }
#pragma unroll
        for (int it = 0; it < 16; ++it) {
            int idx = it * 32 + lane;
            int row = idx >> 5;
            int k   = idx & 31;
            if (k < INF) {
                float xv = x[(size_t)(rowBase + row) * 784 + t * INF + k];
                myx[row * XPITCH + k] = (unsigned char)f32_to_fp8(xv);
            }
        }
        asm volatile("s_wait_dscnt 0" ::: "memory");
        v8i ax;
        {
            const unsigned char* base = myx + colc * XPITCH + hiHalf * 16;
#pragma unroll
            for (int p = 0; p < 4; ++p) {
                uint2 d = *(const uint2*)(base + (p >> 1) * 32 + (p & 1) * 8);
                ax[2*p]   = (int)d.x;
                ax[2*p+1] = (int)d.y;
            }
        }
#pragma unroll
        for (int j = 0; j < 8; ++j) {
            float bh = lbias1n[j * 16 + colc];
            v8f ar = {}, az = {}, ai = {};
            v8f ahn;
#pragma unroll
            for (int i = 0; i < 8; ++i) ahn[i] = bh;
            ar  = wmma64 (ax,  loadB64 (ldsw + W1X_OFF + (0*8+j)*1024, lane), ar);
            az  = wmma64 (ax,  loadB64 (ldsw + W1X_OFF + (1*8+j)*1024, lane), az);
            ai  = wmma64 (ax,  loadB64 (ldsw + W1X_OFF + (2*8+j)*1024, lane), ai);
            ar  = wmma128(ah1, loadB128(ldsw + W1H_OFF + (0*8+j)*2048, lane), ar);
            az  = wmma128(ah1, loadB128(ldsw + W1H_OFF + (1*8+j)*2048, lane), az);
            ahn = wmma128(ah1, loadB128(ldsw + W1H_OFF + (2*8+j)*2048, lane), ahn);
#pragma unroll
            for (int i = 0; i < 8; ++i) {
                float r  = sigm(ar[i]);
                float z  = sigm(az[i]);
                float nn = tanh_(fmaf(r, ahn[i], ai[i]));
                float hp = h1s[j*8+i];
                float h  = fmaxf(fmaf(z, hp - nn, nn), 0.0f);
                h1s[j*8+i] = h;
                int row = i + hiHalf * 8;
                myh[row * HPITCH + j * 16 + colc] = (unsigned char)f32_to_fp8(h);
            }
        }
        asm volatile("s_wait_dscnt 0" ::: "memory");
        {
            const unsigned char* base = myh + colc * HPITCH + hiHalf * 16;
#pragma unroll
            for (int p = 0; p < 8; ++p) {
                int K0 = (p >> 1) * 32 + (p & 1) * 8;
                uint2 d = *(const uint2*)(base + K0);
                ah1[2*p]   = (int)d.x;
                ah1[2*p+1] = (int)d.y;
            }
        }
#pragma unroll
        for (int j = 0; j < 8; ++j) {
            float br = lbias2[      j * 16 + colc];
            float bz = lbias2[128 + j * 16 + colc];
            float bi = lbias2[256 + j * 16 + colc];
            float bh = lbias2[384 + j * 16 + colc];
            v8f ar, az, ai, ahn;
#pragma unroll
            for (int i = 0; i < 8; ++i) { ar[i]=br; az[i]=bz; ai[i]=bi; ahn[i]=bh; }
            ar  = wmma128(ah1, loadB128(ldsw + W2X_OFF + (0*8+j)*2048, lane), ar);
            az  = wmma128(ah1, loadB128(ldsw + W2X_OFF + (1*8+j)*2048, lane), az);
            ai  = wmma128(ah1, loadB128(ldsw + W2X_OFF + (2*8+j)*2048, lane), ai);
            ar  = wmma128(ah2, loadB128(ldsw + W2H_OFF + (0*8+j)*2048, lane), ar);
            az  = wmma128(ah2, loadB128(ldsw + W2H_OFF + (1*8+j)*2048, lane), az);
            ahn = wmma128(ah2, loadB128(ldsw + W2H_OFF + (2*8+j)*2048, lane), ahn);
#pragma unroll
            for (int i = 0; i < 8; ++i) {
                float r  = sigm(ar[i]);
                float z  = sigm(az[i]);
                float nn = tanh_(fmaf(r, ahn[i], ai[i]));
                float hp = h2s[j*8+i];
                float h  = fmaxf(fmaf(z, hp - nn, nn), 0.0f);
                h2s[j*8+i] = h;
                int row = i + hiHalf * 8;
                myh[row * HPITCH + j * 16 + colc] = (unsigned char)f32_to_fp8(h);
            }
        }
        asm volatile("s_wait_dscnt 0" ::: "memory");
        {
            const unsigned char* base = myh + colc * HPITCH + hiHalf * 16;
#pragma unroll
            for (int p = 0; p < 8; ++p) {
                int K0 = (p >> 1) * 32 + (p & 1) * 8;
                uint2 d = *(const uint2*)(base + K0);
                ah2[2*p]   = (int)d.x;
                ah2[2*p+1] = (int)d.y;
            }
        }
    }

    __syncthreads();
    float* ob = (float*)ldsw + wave * (16 * 132);
#pragma unroll
    for (int j = 0; j < 8; ++j)
#pragma unroll
        for (int i = 0; i < 8; ++i) {
            int row = i + hiHalf * 8;
            ob[row * 132 + j * 16 + colc] = h2s[j*8+i];
        }
    asm volatile("s_wait_dscnt 0" ::: "memory");
    for (int idx = lane; idx < 160; idx += 32) {
        int row = idx / 10;
        int o   = idx - row * 10;
        float sum = b_out[o];
        for (int k = 0; k < HID; ++k)
            sum += ob[row * 132 + k] * w_out[o * HID + k];
        out[(size_t)(rowBase + row) * 10 + o] = sum;
    }
}

extern "C" void kernel_launch(void* const* d_in, const int* in_sizes, int n_in,
                              void* d_out, int out_size, void* d_ws, size_t ws_size,
                              hipStream_t stream) {
    (void)n_in; (void)out_size; (void)ws_size;
    const float* x     = (const float*)d_in[0];
    const float* w_ih1 = (const float*)d_in[1];
    const float* w_hh1 = (const float*)d_in[2];
    const float* b_ih1 = (const float*)d_in[3];
    const float* b_hh1 = (const float*)d_in[4];
    const float* w_ih2 = (const float*)d_in[5];
    const float* w_hh2 = (const float*)d_in[6];
    const float* b_ih2 = (const float*)d_in[7];
    const float* b_hh2 = (const float*)d_in[8];
    const float* w_out = (const float*)d_in[9];
    const float* b_out = (const float*)d_in[10];
    unsigned char* ws  = (unsigned char*)d_ws;

    int nBatch = in_sizes[0] / 784;
    int blocks = nBatch / MTILE;

    pack_weights_kernel<<<(WBYTES + 255) / 256, 256, 0, stream>>>(w_ih1, w_hh1, w_ih2, w_hh2,
                                                                  b_ih1, b_hh1, ws);
    gru_main<<<blocks, 256, 0, stream>>>(x, b_ih1, b_hh1, b_ih2, b_hh2,
                                         w_out, b_out, ws, (float*)d_out);
}
